// BlockAttention_11897059410184
// MI455X (gfx1250) — matrix operand order repaired and checked by reading, not compiled, not run on hardware
//
#include <hip/hip_runtime.h>
#include <math.h>

typedef __attribute__((ext_vector_type(16))) __bf16 v16bf;
typedef __attribute__((ext_vector_type(8)))  float  v8f;
typedef __attribute__((ext_vector_type(4)))  float  v4f;

#define BS    32
#define NH    16      // "K" heads in the reference
#define TT    4
#define DD    64
#define NV    8192
#define WAVES 16
#define NTHREADS (WAVES * 32)        // 512
#define HALVES 2                     // vocab split across blocks
#define VPW   (NV / HALVES / WAVES)  // 256 vocab rows per wave
#define NIT   (VPW / 32)             // 8 blocks of 32 vocab rows
#define RSTRIDE (TT * DD)            // 256 floats between vocab rows
#define SLOTS (BS * TT)              // 128 (b,t) problems
#define ACCF  (NH * DD + NH)         // 1040 floats per slot: acc + l

// ---------------- workspace init (d_ws is poisoned before timing) ----------------
__global__ void init_ws_kernel(float* wsf, unsigned long long* wsk) {
    int i = blockIdx.x * blockDim.x + threadIdx.x;
    for (int j = i; j < SLOTS * ACCF; j += gridDim.x * blockDim.x) wsf[j] = 0.0f;
    for (int j = i; j < SLOTS * NH;   j += gridDim.x * blockDim.x) wsk[j] = 0ull;
}

// ---------------- main flash-attention partial kernel ----------------
__global__ __launch_bounds__(NTHREADS, 1)
void block_attn_partial(const float* __restrict__ qg,
                        const float* __restrict__ kg,
                        const float* __restrict__ vg,
                        float* __restrict__ wsf,
                        unsigned long long* __restrict__ wsk)
{
    __shared__ __bf16 p_lds[WAVES * NH * 32];        // 16 KB: per-wave P staging
    __shared__ float  acc_lds[NH * DD];              // 4 KB : block-local O accumulator
    __shared__ float  l_lds[NH];                     // block-local denominators
    __shared__ unsigned long long key_lds[NH];       // block-local argmax keys

    const int tid  = threadIdx.x;
    const int w    = tid >> 5;
    const int lane = tid & 31;
    const int hf   = lane >> 4;       // which 16-lane half of the wave
    const int ln   = lane & 15;

    const int slot = blockIdx.x >> 1;
    const int half = blockIdx.x & 1;
    const int b = slot / TT;
    const int t = slot % TT;

    // ---- init shared merge buffers ----
    for (int i = tid; i < NH * DD; i += NTHREADS) acc_lds[i] = 0.0f;
    if (tid < NH) { l_lds[tid] = 0.0f; key_lds[tid] = 0ull; }
    __syncthreads();

    // ---- base pointers: x[b, row, t, d] with row stride 256 floats ----
    const float* kbase = kg + (((size_t)b * NV) * TT + t) * DD;
    const float* vbase = vg + (((size_t)b * NV) * TT + t) * DD;

    // ---- q as two A-fragments (16x32 bf16 each) ----
    // Scale folds d^-0.5 AND log2(e) so exp2(S) == exp(logit): v_exp_f32 is base-2.
    // A layout: lane -> M = ln ; half h: K = hf*16 + h
    v16bf a0, a1;
    {
        const float qscale = 0.125f * 1.4426950408889634f;
        const float* qp = qg + (((size_t)b * NH + ln) * TT + t) * DD;
        #pragma unroll
        for (int h = 0; h < 16; ++h) {
            int g    = h >> 3;
            int kidx = hf * 16 + h;
            a0[h] = (__bf16)(qp[kidx] * qscale);
            a1[h] = (__bf16)(qp[32 + kidx] * qscale);
        }
    }

    // B-fragment of K for the S-GEMM: lane holds 16 consecutive d values
    auto loadKfrag = [&](const float* rowp) -> v16bf {
        v16bf r;
        const v4f* p4 = (const v4f*)rowp;
        #pragma unroll
        for (int i = 0; i < 4; ++i) {
            v4f f = __builtin_nontemporal_load(p4 + i);
            r[4*i+0] = (__bf16)f.x; r[4*i+1] = (__bf16)f.y;
            r[4*i+2] = (__bf16)f.z; r[4*i+3] = (__bf16)f.w;
        }
        return r;
    };
    // B-fragment of V for the P@V GEMM: lane = d column, halves walk vocab rows
    auto loadVfrag = [&](int vb, int j) -> v16bf {
        const float* p = vbase + (size_t)(vb + hf * 16) * RSTRIDE + j * 16 + ln;
        v16bf r;
        #pragma unroll
        for (int h = 0; h < 16; ++h)
            r[h] = (__bf16)__builtin_nontemporal_load(p + (size_t)h * RSTRIDE);
        return r;
    };

    // ---- per-lane partial state: rows m = r + hf*8, this lane's columns only ----
    float lsum[8], bval[8];
    int   bidx[8];
    v8f   accv[4] = {};   // O accumulator: 4 d-tiles x 8 rows
    #pragma unroll
    for (int r = 0; r < 8; ++r) { lsum[r] = 0.0f; bval[r] = -INFINITY; bidx[r] = 0; }

    const int vstart = half * (NV / HALVES) + w * VPW;
    __bf16* pw = p_lds + (size_t)w * NH * 32;

    // ---- double-buffered K fragments: preload block 0 ----
    v16bf c00 = loadKfrag(kbase + (size_t)(vstart + ln)      * RSTRIDE + hf * 16);
    v16bf c01 = loadKfrag(kbase + (size_t)(vstart + ln)      * RSTRIDE + 32 + hf * 16);
    v16bf c10 = loadKfrag(kbase + (size_t)(vstart + 16 + ln) * RSTRIDE + hf * 16);
    v16bf c11 = loadKfrag(kbase + (size_t)(vstart + 16 + ln) * RSTRIDE + 32 + hf * 16);

    for (int it = 0; it < NIT; ++it) {
        const int vb = vstart + it * 32;
        v16bf b00 = c00, b01 = c01, b10 = c10, b11 = c11;

        // ---- issue next block's K loads + prefetch hints early ----
        if (it + 1 < NIT) {
            const int nb = vb + 32;
            const float* kn = kbase + (size_t)(nb + lane) * RSTRIDE;
            const float* vn = vbase + (size_t)(nb + lane) * RSTRIDE;
            __builtin_prefetch(vn, 0, 0);
            __builtin_prefetch(vn + 32, 0, 0);
            c00 = loadKfrag(kbase + (size_t)(nb + ln)      * RSTRIDE + hf * 16);
            c01 = loadKfrag(kbase + (size_t)(nb + ln)      * RSTRIDE + 32 + hf * 16);
            c10 = loadKfrag(kbase + (size_t)(nb + 16 + ln) * RSTRIDE + hf * 16);
            c11 = loadKfrag(kbase + (size_t)(nb + 16 + ln) * RSTRIDE + 32 + hf * 16);
            __builtin_prefetch(kn + 64 * RSTRIDE, 0, 0);   // distance-2 K prefetch
            __builtin_prefetch(kn + 64 * RSTRIDE + 32, 0, 0);
        }

        // ---- S = q @ K^T for 32 vocab rows (two 16x16 tiles, K=64 split 32+32) ----
        v8f zero = {};
        v8f s0 = __builtin_amdgcn_wmma_f32_16x16x32_bf16(false, a0, false, b00, (short)0, zero, false, false);
        s0     = __builtin_amdgcn_wmma_f32_16x16x32_bf16(false, a1, false, b01, (short)0, s0,   false, false);
        v8f s1 = __builtin_amdgcn_wmma_f32_16x16x32_bf16(false, a0, false, b10, (short)0, zero, false, false);
        s1     = __builtin_amdgcn_wmma_f32_16x16x32_bf16(false, a1, false, b11, (short)0, s1,   false, false);

        // ---- P = exp2(S) == exp(logit); per-lane partial sums + argmax; stage P ----
        #pragma unroll
        for (int r = 0; r < 8; ++r) {
            float v0 = s0[r], v1 = s1[r];
            if (v0 > bval[r]) { bval[r] = v0; bidx[r] = vb + ln; }
            if (v1 > bval[r]) { bval[r] = v1; bidx[r] = vb + 16 + ln; }
            float p0 = exp2f(v0);
            float p1 = exp2f(v1);
            lsum[r] += p0 + p1;
            int m = r + hf * 8;
            pw[m * 32 + ln]      = (__bf16)p0;           // C-layout: lane = vocab col
            pw[m * 32 + 16 + ln] = (__bf16)p1;
        }

        // ---- transpose P to A-layout via LDS, then O += P @ V (K=32) ----
        v16bf pa;
        #pragma unroll
        for (int h = 0; h < 16; ++h) {
            int g  = h >> 3;
            int kv = hf * 16 + h;
            pa[h] = pw[ln * 32 + kv];
        }
        #pragma unroll
        for (int j = 0; j < 4; ++j) {
            v16bf vf = loadVfrag(vb, j);
            accv[j] = __builtin_amdgcn_wmma_f32_16x16x32_bf16(false, pa, false, vf, (short)0, accv[j], false, false);
        }
    }

    // ---- one-time intra-wave reductions across the 16 lanes of each half ----
    #pragma unroll
    for (int r = 0; r < 8; ++r) {
        float ls = lsum[r];
        #pragma unroll
        for (int s = 1; s < 16; s <<= 1) ls += __shfl_xor(ls, s, 32);
        lsum[r] = ls;

        float bv_ = bval[r]; int bi_ = bidx[r];
        #pragma unroll
        for (int s = 1; s < 16; s <<= 1) {
            float ov = __shfl_xor(bv_, s, 32);
            int   oi = __shfl_xor(bi_, s, 32);
            if (ov > bv_ || (ov == bv_ && oi < bi_)) { bv_ = ov; bi_ = oi; }
        }
        bval[r] = bv_; bidx[r] = bi_;
    }

    // ---- merge across waves via LDS atomics (no rescale needed) ----
    #pragma unroll
    for (int r = 0; r < 8; ++r) {
        int m = r + hf * 8;
        if (ln == 0) {
            atomicAdd(&l_lds[m], lsum[r]);
            unsigned ub = __float_as_uint(bval[r]);
            ub = (ub & 0x80000000u) ? ~ub : (ub | 0x80000000u);   // order-preserving key
            unsigned long long key =
                ((unsigned long long)ub << 32) |
                (unsigned long long)(0xFFFFFFFFu - (unsigned)bidx[r]); // ties -> lowest idx
            atomicMax(&key_lds[m], key);
        }
        #pragma unroll
        for (int j = 0; j < 4; ++j)
            atomicAdd(&acc_lds[m * DD + j * 16 + ln], accv[j][r]);
    }
    __syncthreads();

    // ---- merge this block's partials into the global workspace ----
    float* accw = wsf + (size_t)slot * ACCF;
    for (int i = tid; i < NH * DD; i += NTHREADS)
        atomicAdd(&accw[i], acc_lds[i]);
    if (tid < NH) {
        atomicAdd(&accw[NH * DD + tid], l_lds[tid]);
        atomicMax(&wsk[(size_t)slot * NH + tid], key_lds[tid]);
    }
}

// ---------------- finalize: normalize + emit out/tokens ----------------
__global__ __launch_bounds__(NTHREADS)
void finalize_kernel(const float* __restrict__ wsf,
                     const unsigned long long* __restrict__ wsk,
                     float* __restrict__ outg,
                     float* __restrict__ tokg)
{
    const int slot = blockIdx.x;
    const int b = slot / TT;
    const int t = slot % TT;
    const float* accw = wsf + (size_t)slot * ACCF;
    const int tid = threadIdx.x;

    for (int i = tid; i < NH * DD; i += NTHREADS) {
        int m = i >> 6, d = i & 63;
        float o = accw[i] / accw[NH * DD + m];
        outg[(((size_t)b * NH + m) * TT + t) * DD + d] = o;
    }
    if (tid < NH) {
        unsigned idx = 0xFFFFFFFFu - (unsigned)(wsk[(size_t)slot * NH + tid] & 0xFFFFFFFFull);
        tokg[((size_t)b * NH + tid) * TT + t] = (float)idx;
    }
}

extern "C" void kernel_launch(void* const* d_in, const int* in_sizes, int n_in,
                              void* d_out, int out_size, void* d_ws, size_t ws_size,
                              hipStream_t stream) {
    (void)in_sizes; (void)n_in; (void)out_size; (void)ws_size;
    const float* q = (const float*)d_in[0];
    const float* k = (const float*)d_in[1];
    const float* v = (const float*)d_in[2];
    float* out = (float*)d_out;
    float* tok = out + (size_t)BS * NH * TT * DD;   // tokens follow the O tensor

    float* wsf = (float*)d_ws;                                    // 128*1040 floats
    unsigned long long* wsk = (unsigned long long*)(wsf + (size_t)SLOTS * ACCF);

    hipLaunchKernelGGL(init_ws_kernel, dim3(64), dim3(512), 0, stream, wsf, wsk);
    hipLaunchKernelGGL(block_attn_partial, dim3(SLOTS * HALVES), dim3(NTHREADS), 0, stream,
                       q, k, v, wsf, wsk);
    hipLaunchKernelGGL(finalize_kernel, dim3(SLOTS), dim3(NTHREADS), 0, stream,
                       wsf, wsk, out, tok);
}
